// GINConvNet_2860448219741
// MI455X (gfx1250) — matrix operand order repaired and checked by reading, not compiled, not run on hardware
//
#include <hip/hip_runtime.h>

#define N_NODES  100000
#define N_PAD    100032   // multiple of 64 (16 rows * 4 waves) -> guard-free tiles
#define N_EDGES  1600000
#define N_GRAPHS 2048
#define NFEAT    78
#define K0_PAD   96       // layer-0 K padded to 3 WMMA k-tiles, zero-filled
#define DIM      32
#define LAYERS   5
#define OUT_DIM  128
#define BN_EPS   1e-5f
#define MLP_WAVES 4

typedef __attribute__((ext_vector_type(16))) __bf16 v16bf;
typedef __attribute__((ext_vector_type(8)))  float  v8f;

// ---- fp32 -> bf16 (round-to-nearest-even), two floats packed per u32 ----
__device__ __forceinline__ unsigned rne_hi(float f) {
  union { float f; unsigned u; } v; v.f = f;
  return (v.u + 0x7FFFu + ((v.u >> 16) & 1u)) & 0xFFFF0000u;
}
__device__ __forceinline__ unsigned rne_lo(float f) {
  union { float f; unsigned u; } v; v.f = f;
  return ((v.u + 0x7FFFu + ((v.u >> 16) & 1u)) >> 16) & 0xFFFFu;
}
__device__ __forceinline__ __bf16 f2bf(float f) {
  unsigned short s = (unsigned short)rne_lo(f);
  __bf16 b; __builtin_memcpy(&b, &s, sizeof(b)); return b;
}
__device__ __forceinline__ v16bf cvt16(const float* r) {
  union { v16bf v; unsigned u[8]; } o;
#pragma unroll
  for (int j = 0; j < 8; ++j) o.u[j] = rne_lo(r[2 * j]) | rne_hi(r[2 * j + 1]);
  return o.v;
}
__device__ __forceinline__ v8f wmma_bf16(v16bf a, v16bf b, v8f c) {
  return __builtin_amdgcn_wmma_f32_16x16x32_bf16(false, a, false, b, (short)0, c,
                                                 false, false);
}

// ---------------- setup / elementwise kernels ----------------

__global__ void k_zero(float* __restrict__ p, int n) {
  int i = blockIdx.x * blockDim.x + threadIdx.x;
  if (i < n) p[i] = 0.f;
}

// agg0[n,96] = x[n,78] zero-padded (self term)
__global__ void k_init_agg0(const float* __restrict__ x, float* __restrict__ agg) {
  int i = blockIdx.x * blockDim.x + threadIdx.x;
  if (i >= N_NODES * K0_PAD) return;
  int n = i / K0_PAD, f = i - n * K0_PAD;
  agg[i] = (f < NFEAT) ? x[n * NFEAT + f] : 0.f;
}

__global__ void k_copy4(const float4* __restrict__ src, float4* __restrict__ dst, int n4) {
  int i = blockIdx.x * blockDim.x + threadIdx.x;
  if (i < n4) dst[i] = src[i];
}

// ---- wave-per-edge scatter: agg[dst] += feat[src]; edge indices scalarized ----
template <int CHUNKS, int FS_SRC, int FS_AGG, int NFV>
__global__ void k_scatter(const float* __restrict__ feat, float* __restrict__ agg,
                          const int* __restrict__ esrc, const int* __restrict__ edst) {
  int w = blockIdx.x * 8 + (threadIdx.x >> 5);   // global wave id (256 thr = 8 waves)
  int lane = threadIdx.x & 31;
  int e = w / CHUNKS;
  int c = w - e * CHUNKS;
  e = __builtin_amdgcn_readfirstlane(e);         // wave-uniform -> SGPR
  c = __builtin_amdgcn_readfirstlane(c);
  if (e >= N_EDGES) return;
  int se = esrc[e];                              // scalar loads
  int de = edst[e];
  int f = c * 32 + lane;
  if (f < NFV)
    atomicAdd(&agg[(size_t)de * FS_AGG + f], feat[(size_t)se * FS_SRC + f]);
}

// Pre-pack fp32 weights [K, ldw] into bf16 WMMA B-layout:
// tile t = kt*ntn + nt; lane L holds elems e with K = kt*32 + (e&7) + 8*(L>>4) + 16*(e>>3),
// col = nt*16 + (L&15). Stored as out[(t*32+L)*16 + e].
__global__ void k_pack_b(const float* __restrict__ W, int ldw, int kvalid,
                         int ktiles, int ntn, __bf16* __restrict__ out) {
  int i = blockIdx.x * blockDim.x + threadIdx.x;
  if (i >= ktiles * ntn * 512) return;
  int e = i & 15;
  int lane = (i >> 4) & 31;
  int t = i >> 9;
  int kt = t / ntn, nt = t - kt * ntn;
  int k = kt * 32 + (e & 7) + 8 * (lane >> 4) + 16 * (e >> 3);
  int col = nt * 16 + (lane & 15);
  out[i] = f2bf((k < kvalid) ? W[k * ldw + col] : 0.f);
}

// ---------------- fused GIN MLP + BN stats ----------------
template <int KTILES, int LDA>
__global__ void __launch_bounds__(MLP_WAVES * 32)
k_gin_mlp(const float* __restrict__ A,
          const __bf16* __restrict__ W1p, const float* __restrict__ b1v,
          const __bf16* __restrict__ W2p, const float* __restrict__ b2v,
          float* __restrict__ hout, float* __restrict__ gsum, float* __restrict__ gsq) {
  __shared__ __align__(16) __bf16 tile[MLP_WAVES][16][DIM];  // intermediate in bf16
  __shared__ float ssum[DIM], ssq[DIM];

  const int tid  = threadIdx.x;
  const int wave = tid >> 5, lane = tid & 31;
  const int m = lane & 15, hf = lane >> 4;

  if (tid < DIM) { ssum[tid] = 0.f; ssq[tid] = 0.f; }
  __syncthreads();

  const int mt = blockIdx.x * MLP_WAVES + wave;
  const float* Abase = A + (size_t)(mt * 16 + m) * LDA + 8 * hf;

  // ---- GEMM1: guard-free vector loads, pre-packed bf16 B ----
  v8f c0 = {}, c1 = {};
#pragma unroll
  for (int kt = 0; kt < KTILES; ++kt) {
    if (KTILES > 1 && kt + 1 < KTILES)
      __builtin_prefetch(Abase + (kt + 1) * 32, 0, 0);
    float r[16];
    const float4* ap = (const float4*)(Abase + kt * 32);
    *(float4*)(r + 0)  = ap[0];  // K = kt*32 + 8*hf + 0..3
    *(float4*)(r + 4)  = ap[1];
    *(float4*)(r + 8)  = ap[4];  // K = kt*32 + 16 + 8*hf + 0..3
    *(float4*)(r + 12) = ap[5];
    v16bf a  = cvt16(r);
    v16bf b0 = *(const v16bf*)(W1p + (size_t)((kt * 2 + 0) * 32 + lane) * 16);
    v16bf b1 = *(const v16bf*)(W1p + (size_t)((kt * 2 + 1) * 32 + lane) * 16);
    c0 = wmma_bf16(a, b0, c0);
    c1 = wmma_bf16(a, b1, c1);
  }

  // ---- bias + ReLU, stage intermediate tile as bf16 (C-layout -> row-major) ----
  {
    float bb0 = b1v[m], bb1 = b1v[16 + m];
#pragma unroll
    for (int r_ = 0; r_ < 8; ++r_) {
      int M = r_ + 8 * hf;
      tile[wave][M][m]      = f2bf(fmaxf(c0[r_] + bb0, 0.f));
      tile[wave][M][16 + m] = f2bf(fmaxf(c1[r_] + bb1, 0.f));
    }
  }
  __syncthreads();

  // ---- GEMM2: A operand = two 16B LDS loads, zero conversions ----
  v8f d0 = {}, d1 = {};
  {
    union { v16bf v; uint4 q[2]; } ua;
    const __bf16* lp = &tile[wave][m][0];
    ua.q[0] = *(const uint4*)(lp + 8 * hf);        // elems 0..7:  K = 8*hf + 0..7
    ua.q[1] = *(const uint4*)(lp + 16 + 8 * hf);   // elems 8..15: K = 16 + 8*hf + 0..7
    v16bf a2 = ua.v;
    v16bf b0 = *(const v16bf*)(W2p + (size_t)(0 * 32 + lane) * 16);
    v16bf b1 = *(const v16bf*)(W2p + (size_t)(1 * 32 + lane) * 16);
    d0 = wmma_bf16(a2, b0, d0);
    d1 = wmma_bf16(a2, b1, d1);
  }

  // ---- bias + ReLU + unguarded store (h padded) + BN partial sums ----
  {
    float bb0 = b2v[m], bb1 = b2v[16 + m];
    float s0 = 0.f, q0 = 0.f, s1 = 0.f, q1 = 0.f;
#pragma unroll
    for (int r_ = 0; r_ < 8; ++r_) {
      int row = mt * 16 + r_ + 8 * hf;
      float v0 = fmaxf(d0[r_] + bb0, 0.f);
      float v1 = fmaxf(d1[r_] + bb1, 0.f);
      hout[(size_t)row * DIM + m]      = v0;
      hout[(size_t)row * DIM + 16 + m] = v1;
      if (row < N_NODES) { s0 += v0; q0 += v0 * v0; s1 += v1; q1 += v1 * v1; }
    }
    atomicAdd(&ssum[m], s0);      atomicAdd(&ssq[m], q0);
    atomicAdd(&ssum[16 + m], s1); atomicAdd(&ssq[16 + m], q1);
  }
  __syncthreads();
  if (tid < DIM) {
    atomicAdd(&gsum[tid], ssum[tid]);
    atomicAdd(&gsq[tid],  ssq[tid]);
  }
}

// ---------------- BatchNorm: fold stats into per-feature scale/shift ----------------
__global__ void k_bnprep(const float* __restrict__ gsum, const float* __restrict__ gsq,
                         const float* __restrict__ gamma, const float* __restrict__ beta,
                         float* __restrict__ coef) {
  int f = threadIdx.x;
  if (f >= DIM) return;
  const float inv_n = 1.0f / (float)N_NODES;
  float mu  = gsum[f] * inv_n;
  float var = gsq[f] * inv_n - mu * mu;
  float sc  = gamma[f] * rsqrtf(var + BN_EPS);
  coef[f]       = sc;
  coef[DIM + f] = beta[f] - mu * sc;
}

__global__ void k_bn4(float4* __restrict__ h, const float* __restrict__ coef) {
  int i = blockIdx.x * blockDim.x + threadIdx.x;
  if (i >= N_NODES * (DIM / 4)) return;
  int f0 = (i & 7) * 4;
  float4 v = h[i];
  v.x = v.x * coef[f0 + 0] + coef[DIM + f0 + 0];
  v.y = v.y * coef[f0 + 1] + coef[DIM + f0 + 1];
  v.z = v.z * coef[f0 + 2] + coef[DIM + f0 + 2];
  v.w = v.w * coef[f0 + 3] + coef[DIM + f0 + 3];
  h[i] = v;
}

// ---------------- global_add_pool: wave-per-node, scalarized batch index ----------------
__global__ void k_pool(const float* __restrict__ h, const int* __restrict__ batch,
                       float* __restrict__ pooled) {
  int w = blockIdx.x * 8 + (threadIdx.x >> 5);
  int lane = threadIdx.x & 31;
  int n = __builtin_amdgcn_readfirstlane(w);
  if (n >= N_NODES) return;
  int g = batch[n];                               // scalar load
  atomicAdd(&pooled[g * DIM + lane], h[(size_t)n * DIM + lane]);
}

// ---------------- final FC: out = ReLU(pooled @ Wfc + bfc), guard-free ----------------
__global__ void __launch_bounds__(128)
k_fc(const float* __restrict__ P, const __bf16* __restrict__ Wp,
     const float* __restrict__ bfc, float* __restrict__ out) {
  const int tid = threadIdx.x, wave = tid >> 5, lane = tid & 31;
  const int m = lane & 15, hf = lane >> 4;
  const int mt = blockIdx.x * 4 + wave;

  float r[16];
  const float* pb = P + (size_t)(mt * 16 + m) * DIM + 8 * hf;
  *(float4*)(r + 0)  = *(const float4*)(pb + 0);
  *(float4*)(r + 4)  = *(const float4*)(pb + 4);
  *(float4*)(r + 8)  = *(const float4*)(pb + 16);
  *(float4*)(r + 12) = *(const float4*)(pb + 20);
  v16bf a = cvt16(r);

#pragma unroll
  for (int nt = 0; nt < OUT_DIM / 16; ++nt) {
    v16bf b = *(const v16bf*)(Wp + (size_t)(nt * 32 + lane) * 16);
    v8f c = {};
    c = wmma_bf16(a, b, c);
    float bb = bfc[nt * 16 + m];
#pragma unroll
    for (int r_ = 0; r_ < 8; ++r_)
      out[(size_t)(mt * 16 + r_ + 8 * hf) * OUT_DIM + nt * 16 + m] =
          fmaxf(c[r_] + bb, 0.f);
  }
}

// ---------------- host launcher ----------------
extern "C" void kernel_launch(void* const* d_in, const int* in_sizes, int n_in,
                              void* d_out, int out_size, void* d_ws, size_t ws_size,
                              hipStream_t stream) {
  const float* x       = (const float*)d_in[0];
  const int*   esrc    = (const int*)d_in[1];
  const int*   edst    = esrc + N_EDGES;
  const int*   batch   = (const int*)d_in[2];
  const float* W1_0    = (const float*)d_in[3];
  const float* W1_rest = (const float*)d_in[4];
  const float* b1      = (const float*)d_in[5];
  const float* W2      = (const float*)d_in[6];
  const float* b2      = (const float*)d_in[7];
  const float* gamma   = (const float*)d_in[8];
  const float* beta    = (const float*)d_in[9];
  const float* Wfc     = (const float*)d_in[10];
  const float* bfc     = (const float*)d_in[11];
  float*       out     = (float*)d_out;

  // ws layout (fp32): agg (N_PAD*96, reused at stride 32), h (N_PAD*32),
  // stats (sum32|sq32|coef64), pooled (G*32), bf16 packed weights (~32 KB).
  float*  agg    = (float*)d_ws;
  float*  h      = agg + (size_t)N_PAD * K0_PAD;
  float*  stats  = h + (size_t)N_PAD * DIM;     // 128 floats
  float*  coef   = stats + 64;
  float*  pooled = stats + 128;
  __bf16* wpk    = (__bf16*)(pooled + N_GRAPHS * DIM);
  __bf16* W1p0  = wpk;                     // 6 tiles  * 512
  __bf16* W1pL  = W1p0 + 6 * 512;          // 4 layers * 2 tiles * 512
  __bf16* W2pL  = W1pL + 4 * 1024;         // 5 layers * 2 tiles * 512
  __bf16* Wfcp  = W2pL + 5 * 1024;         // 8 tiles  * 512

  const int TPB = 256;
  const int mlp_grid = N_PAD / (16 * MLP_WAVES);  // 1563, exact
  const int bn_grid  = (N_NODES * (DIM / 4) + TPB - 1) / TPB;

  // ---- pre-pack all weights into bf16 WMMA B-layout ----
  k_pack_b<<<(3 * 2 * 512 + TPB - 1) / TPB, TPB, 0, stream>>>(W1_0, DIM, NFEAT, 3, 2, W1p0);
  for (int l = 1; l < LAYERS; ++l)
    k_pack_b<<<4, TPB, 0, stream>>>(
        W1_rest + (size_t)(l - 1) * DIM * DIM, DIM, DIM, 1, 2, W1pL + (l - 1) * 1024);
  for (int l = 0; l < LAYERS; ++l)
    k_pack_b<<<4, TPB, 0, stream>>>(
        W2 + (size_t)l * DIM * DIM, DIM, DIM, 1, 2, W2pL + l * 1024);
  k_pack_b<<<16, TPB, 0, stream>>>(Wfc, OUT_DIM, DIM, 1, 8, Wfcp);

  // ---- layer 0 (K = 78 -> 96) ----
  k_init_agg0<<<(N_NODES * K0_PAD + TPB - 1) / TPB, TPB, 0, stream>>>(x, agg);
  k_scatter<3, NFEAT, K0_PAD, NFEAT>
      <<<(N_EDGES * 3 + 7) / 8, TPB, 0, stream>>>(x, agg, esrc, edst);
  k_zero<<<1, 64, 0, stream>>>(stats, 64);
  k_gin_mlp<3, K0_PAD><<<mlp_grid, MLP_WAVES * 32, 0, stream>>>(
      agg, W1p0, b1, W2pL, b2, h, stats, stats + 32);
  k_bnprep<<<1, 32, 0, stream>>>(stats, stats + 32, gamma, beta, coef);
  k_bn4<<<bn_grid, TPB, 0, stream>>>((float4*)h, coef);

  // ---- layers 1..4 (K = 32, single k-tile) ----
  for (int l = 1; l < LAYERS; ++l) {
    k_copy4<<<(N_NODES * DIM / 4 + TPB - 1) / TPB, TPB, 0, stream>>>(
        (const float4*)h, (float4*)agg, N_NODES * DIM / 4);
    k_scatter<1, DIM, DIM, DIM>
        <<<(N_EDGES + 7) / 8, TPB, 0, stream>>>(h, agg, esrc, edst);
    k_zero<<<1, 64, 0, stream>>>(stats, 64);
    k_gin_mlp<1, DIM><<<mlp_grid, MLP_WAVES * 32, 0, stream>>>(
        agg, W1pL + (size_t)(l - 1) * 1024, b1 + l * DIM,
        W2pL + (size_t)l * 1024, b2 + l * DIM, h, stats, stats + 32);
    k_bnprep<<<1, 32, 0, stream>>>(stats, stats + 32, gamma + l * DIM,
                                   beta + l * DIM, coef);
    k_bn4<<<bn_grid, TPB, 0, stream>>>((float4*)h, coef);
  }

  // ---- pooling + FC ----
  k_zero<<<(N_GRAPHS * DIM + TPB - 1) / TPB, TPB, 0, stream>>>(pooled, N_GRAPHS * DIM);
  k_pool<<<(N_NODES + 7) / 8, TPB, 0, stream>>>(h, batch, pooled);
  k_fc<<<N_GRAPHS / 64, 128, 0, stream>>>(pooled, Wfcp, bfc, out);
}
